// MultiHeadAttention_39135742002005
// MI455X (gfx1250) — matrix operand order repaired and checked by reading, not compiled, not run on hardware
//
#include <hip/hip_runtime.h>

#define BB 8
#define TT 1024
#define CC 768
#define HH 12
#define DD 64
#define BT (BB * TT) /* 8192 */

typedef __bf16 bf16_t;
typedef __attribute__((ext_vector_type(16))) __bf16 v16bf;
typedef __attribute__((ext_vector_type(8))) float v8f;
typedef __attribute__((ext_vector_type(4))) unsigned int u32x4;

union Frag16 {
    v16bf v;
    u32x4 q[2];
};

static __device__ __forceinline__ v16bf load_frag(const bf16_t* p0, const bf16_t* p1) {
    Frag16 f;
    f.q[0] = *(const u32x4*)p0;
    f.q[1] = *(const u32x4*)p1;
    return f.v;
}

static __device__ __forceinline__ v8f wmma_bf16(v16bf a, v16bf b, v8f c) {
    return __builtin_amdgcn_wmma_f32_16x16x32_bf16(false, a, false, b, (short)0, c, false, false);
}

// ---------------- conversion kernels ----------------

__global__ void cvt_f32_to_bf16(const float* __restrict__ in, bf16_t* __restrict__ out, int n) {
    int i = blockIdx.x * 256 + threadIdx.x;
    if (i < n) out[i] = (bf16_t)in[i];
}

// W is [K=CC][N=CC] row-major (torch Linear W^T, i.e. [in,out]); write Wt[N][K] bf16
__global__ void transpose_w_bf16(const float* __restrict__ W, bf16_t* __restrict__ Wt) {
    int i = blockIdx.x * 256 + threadIdx.x;
    if (i >= CC * CC) return;
    int k = i / CC, n = i % CC;
    Wt[(size_t)n * CC + k] = (bf16_t)W[i];
}

// ---------------- QKV projection GEMM ----------------
// Each wave: 32x64 output tile. Per K-step: load phase (2 A-frags + 4 B-frags)
// fully separated from compute phase (8 WMMAs) so the scheduler can overlap
// in-flight loads with matrix issue using partial loadcnt waits.
//   vmode==0: out[((b*H+h)*T + t)*D + d]   (Q, K layout)
//   vmode==1: out[((b*H+h)*D + d)*T + t]   (V transposed for PV matmul)
__global__ __launch_bounds__(128) void qkv_gemm(const bf16_t* __restrict__ xb,
                                                const bf16_t* __restrict__ Wt,
                                                const float* __restrict__ bias,
                                                bf16_t* __restrict__ out, int vmode) {
    int w = threadIdx.x >> 5, lane = threadIdx.x & 31;
    int mrow = lane & 15, kh = lane >> 4;
    int rowBase = blockIdx.x * 32;
    int colBase = (blockIdx.y * 4 + w) * 64;

    v8f acc[2][4] = {};
    const bf16_t* arow0 = xb + (size_t)(rowBase + mrow) * CC;
    const bf16_t* arow1 = xb + (size_t)(rowBase + 16 + mrow) * CC;
    const bf16_t* brow = Wt + (size_t)(colBase + mrow) * CC;

    for (int k0 = 0; k0 < CC; k0 += 32) {
        // ---- load phase ----
        v16bf a0 = load_frag(arow0 + k0 + kh * 16, arow0 + k0 + kh * 16 + 8);
        v16bf a1 = load_frag(arow1 + k0 + kh * 16, arow1 + k0 + kh * 16 + 8);
        v16bf bfr[4];
#pragma unroll
        for (int f = 0; f < 4; ++f) {
            const bf16_t* bp = brow + (size_t)(f * 16) * CC + k0 + kh * 16;
            bfr[f] = load_frag(bp, bp + 8);
        }
        // ---- compute phase ----
#pragma unroll
        for (int f = 0; f < 4; ++f) {
            acc[0][f] = wmma_bf16(a0, bfr[f], acc[0][f]);
            acc[1][f] = wmma_bf16(a1, bfr[f], acc[1][f]);
        }
    }

#pragma unroll
    for (int u = 0; u < 2; ++u) {
#pragma unroll
        for (int f = 0; f < 4; ++f) {
            int col = colBase + f * 16 + mrow;
            float bv = bias[col];
            int h = col >> 6, d = col & 63;
#pragma unroll
            for (int r = 0; r < 8; ++r) {
                int row = rowBase + u * 16 + kh * 8 + r;
                int b = row >> 10, t = row & 1023;
                float val = acc[u][f][r] + bv;
                size_t idx;
                if (vmode)
                    idx = ((size_t)(b * HH + h) * DD + d) * TT + t;
                else
                    idx = ((size_t)(b * HH + h) * TT + t) * DD + d;
                out[idx] = (bf16_t)val;
            }
        }
    }
}

// ---------------- flash attention ----------------
// Q,K: [B*H][T][D] bf16; Vt: [B*H][D][T] bf16; Y: [B][T][C] bf16
__global__ __launch_bounds__(128) void attn_kernel(const bf16_t* __restrict__ Q,
                                                   const bf16_t* __restrict__ K,
                                                   const bf16_t* __restrict__ Vt,
                                                   bf16_t* __restrict__ Y) {
    __shared__ bf16_t Pbuf[4][16][32];
    int w = threadIdx.x >> 5, lane = threadIdx.x & 31;
    int mrow = lane & 15, kh = lane >> 4;
    int bh = blockIdx.y;
    int b = bh / HH, h = bh % HH;
    int q0 = (blockIdx.x * 4 + w) * 16;

    const bf16_t* Qp = Q + (size_t)bh * TT * DD;
    const bf16_t* Kp = K + (size_t)bh * TT * DD;
    const bf16_t* Vp = Vt + (size_t)bh * DD * TT;
    bf16_t(*P)[32] = Pbuf[w];

    // preload the two Q A-fragments (d = 0..31, 32..63); reused across all key tiles
    v16bf qa[2];
#pragma unroll
    for (int i = 0; i < 2; ++i) {
        const bf16_t* qp = Qp + (size_t)(q0 + mrow) * DD + i * 32;
        qa[i] = load_frag(qp + kh * 16, qp + kh * 16 + 8);
    }

    v8f o[4] = {};
    float mr[8], lr[8];
#pragma unroll
    for (int r = 0; r < 8; ++r) {
        mr[r] = -1e30f;
        lr[r] = 0.0f;
    }
    const float scale = 0.125f; // 1/sqrt(64)

    for (int n0 = 0; n0 < TT; n0 += 32) {
        // ---- load all 4 K B-frags, then 4 score WMMAs ----
        v16bf kb[4]; // [i][half]: i = d-half, half = key group
#pragma unroll
        for (int i = 0; i < 2; ++i) {
            const bf16_t* kp0 = Kp + (size_t)(n0 + mrow) * DD + i * 32 + kh * 16;
            const bf16_t* kp1 = Kp + (size_t)(n0 + 16 + mrow) * DD + i * 32 + kh * 16;
            kb[i * 2 + 0] = load_frag(kp0, kp0 + 8);
            kb[i * 2 + 1] = load_frag(kp1, kp1 + 8);
        }
        v8f s0 = {}, s1 = {};
#pragma unroll
        for (int i = 0; i < 2; ++i) {
            s0 = wmma_bf16(qa[i], kb[i * 2 + 0], s0);
            s1 = wmma_bf16(qa[i], kb[i * 2 + 1], s1);
        }

        // online softmax (row r lives in 16 lanes of one half-wave -> xor 1,2,4,8)
        float alpha[8];
#pragma unroll
        for (int r = 0; r < 8; ++r) {
            float v0 = s0[r] * scale, v1 = s1[r] * scale;
            float mx = fmaxf(v0, v1);
            mx = fmaxf(mx, __shfl_xor(mx, 1, 32));
            mx = fmaxf(mx, __shfl_xor(mx, 2, 32));
            mx = fmaxf(mx, __shfl_xor(mx, 4, 32));
            mx = fmaxf(mx, __shfl_xor(mx, 8, 32));
            float mn = fmaxf(mr[r], mx);
            float a = __expf(mr[r] - mn);
            float p0 = __expf(v0 - mn);
            float p1 = __expf(v1 - mn);
            float rs = p0 + p1;
            rs += __shfl_xor(rs, 1, 32);
            rs += __shfl_xor(rs, 2, 32);
            rs += __shfl_xor(rs, 4, 32);
            rs += __shfl_xor(rs, 8, 32);
            lr[r] = lr[r] * a + rs;
            mr[r] = mn;
            alpha[r] = a;
            // stage P to LDS in row-major to re-layout C-format -> A-format
            P[kh * 8 + r][mrow] = (bf16_t)p0;
            P[kh * 8 + r][16 + mrow] = (bf16_t)p1;
        }

        // ---- load all 4 V B-frags while softmax math retires ----
        v16bf vb[4];
#pragma unroll
        for (int g = 0; g < 4; ++g) {
            const bf16_t* vp = Vp + (size_t)(g * 16 + mrow) * TT + n0 + kh * 16;
            vb[g] = load_frag(vp, vp + 8);
        }

#pragma unroll
        for (int g = 0; g < 4; ++g)
#pragma unroll
            for (int r = 0; r < 8; ++r) o[g][r] *= alpha[r];

        asm volatile("s_wait_dscnt 0x0" ::: "memory");
        v16bf pa = load_frag(&P[mrow][kh * 16], &P[mrow][kh * 16 + 8]);

        // O += P @ V
#pragma unroll
        for (int g = 0; g < 4; ++g) o[g] = wmma_bf16(pa, vb[g], o[g]);
    }

    float inv[8];
#pragma unroll
    for (int r = 0; r < 8; ++r) inv[r] = 1.0f / lr[r];
#pragma unroll
    for (int g = 0; g < 4; ++g) {
        int c = h * 64 + g * 16 + mrow;
#pragma unroll
        for (int r = 0; r < 8; ++r) {
            int t = q0 + kh * 8 + r;
            Y[((size_t)b * TT + t) * CC + c] = (bf16_t)(o[g][r] * inv[r]);
        }
    }
}

// ---------------- output projection ----------------
__global__ __launch_bounds__(128) void proj_gemm(const bf16_t* __restrict__ yb,
                                                 const bf16_t* __restrict__ Wt,
                                                 const float* __restrict__ bias,
                                                 float* __restrict__ out) {
    int w = threadIdx.x >> 5, lane = threadIdx.x & 31;
    int mrow = lane & 15, kh = lane >> 4;
    int rowBase = blockIdx.x * 32;
    int colBase = (blockIdx.y * 4 + w) * 64;

    v8f acc[2][4] = {};
    const bf16_t* arow0 = yb + (size_t)(rowBase + mrow) * CC;
    const bf16_t* arow1 = yb + (size_t)(rowBase + 16 + mrow) * CC;
    const bf16_t* brow = Wt + (size_t)(colBase + mrow) * CC;

    for (int k0 = 0; k0 < CC; k0 += 32) {
        v16bf a0 = load_frag(arow0 + k0 + kh * 16, arow0 + k0 + kh * 16 + 8);
        v16bf a1 = load_frag(arow1 + k0 + kh * 16, arow1 + k0 + kh * 16 + 8);
        v16bf bfr[4];
#pragma unroll
        for (int f = 0; f < 4; ++f) {
            const bf16_t* bp = brow + (size_t)(f * 16) * CC + k0 + kh * 16;
            bfr[f] = load_frag(bp, bp + 8);
        }
#pragma unroll
        for (int f = 0; f < 4; ++f) {
            acc[0][f] = wmma_bf16(a0, bfr[f], acc[0][f]);
            acc[1][f] = wmma_bf16(a1, bfr[f], acc[1][f]);
        }
    }

#pragma unroll
    for (int u = 0; u < 2; ++u) {
#pragma unroll
        for (int f = 0; f < 4; ++f) {
            int col = colBase + f * 16 + mrow;
            float bv = bias[col];
#pragma unroll
            for (int r = 0; r < 8; ++r) {
                int row = rowBase + u * 16 + kh * 8 + r;
                out[(size_t)row * CC + col] = acc[u][f][r] + bv;
            }
        }
    }
}

extern "C" void kernel_launch(void* const* d_in, const int* in_sizes, int n_in,
                              void* d_out, int out_size, void* d_ws, size_t ws_size,
                              hipStream_t stream) {
    const float* x = (const float*)d_in[0];
    const float* Wq = (const float*)d_in[1];
    const float* bq = (const float*)d_in[2];
    const float* Wk = (const float*)d_in[3];
    const float* bk = (const float*)d_in[4];
    const float* Wv = (const float*)d_in[5];
    const float* bv = (const float*)d_in[6];
    const float* Wp = (const float*)d_in[7];
    const float* bp = (const float*)d_in[8];
    float* out = (float*)d_out;

    char* ws = (char*)d_ws;
    size_t off = 0;
    bf16_t* xb = (bf16_t*)(ws + off);  off += (size_t)BT * CC * 2;
    bf16_t* Wqt = (bf16_t*)(ws + off); off += (size_t)CC * CC * 2;
    bf16_t* Wkt = (bf16_t*)(ws + off); off += (size_t)CC * CC * 2;
    bf16_t* Wvt = (bf16_t*)(ws + off); off += (size_t)CC * CC * 2;
    bf16_t* Wpt = (bf16_t*)(ws + off); off += (size_t)CC * CC * 2;
    bf16_t* Qb = (bf16_t*)(ws + off);  off += (size_t)BT * CC * 2;
    bf16_t* Kb = (bf16_t*)(ws + off);  off += (size_t)BT * CC * 2;
    bf16_t* Vtb = (bf16_t*)(ws + off); off += (size_t)BT * CC * 2;
    bf16_t* Yb = (bf16_t*)(ws + off);  off += (size_t)BT * CC * 2;

    // 1) convert activations + weights to bf16 (weights transposed for B-operand loads)
    cvt_f32_to_bf16<<<(BT * CC) / 256, 256, 0, stream>>>(x, xb, BT * CC);
    transpose_w_bf16<<<(CC * CC) / 256, 256, 0, stream>>>(Wq, Wqt);
    transpose_w_bf16<<<(CC * CC) / 256, 256, 0, stream>>>(Wk, Wkt);
    transpose_w_bf16<<<(CC * CC) / 256, 256, 0, stream>>>(Wv, Wvt);
    transpose_w_bf16<<<(CC * CC) / 256, 256, 0, stream>>>(Wp, Wpt);

    // 2) QKV projections (V stored transposed per head); 32-row tiles per wave
    dim3 ggrid(BT / 32, 3);
    qkv_gemm<<<ggrid, 128, 0, stream>>>(xb, Wqt, bq, Qb, 0);
    qkv_gemm<<<ggrid, 128, 0, stream>>>(xb, Wkt, bk, Kb, 0);
    qkv_gemm<<<ggrid, 128, 0, stream>>>(xb, Wvt, bv, Vtb, 1);

    // 3) attention: one wave per 16 query rows, 96 (b,h) pairs
    attn_kernel<<<dim3(TT / 64, BB * HH), 128, 0, stream>>>(Qb, Kb, Vtb, Yb);

    // 4) output projection to fp32
    proj_gemm<<<ggrid, 128, 0, stream>>>(Yb, Wpt, bp, out);
}
